// AdaptiveLoss_25400436589246
// MI455X (gfx1250) — compile-verified, matrix operand order checked by reading, not run on hardware
//
#include <hip/hip_runtime.h>
#include <hip/hip_bf16.h>
#include <math.h>

typedef __attribute__((ext_vector_type(2)))  float    v2f;
typedef __attribute__((ext_vector_type(8)))  float    v8f;
typedef __attribute__((ext_vector_type(16))) _Float16 v16h;

#define C_DIM 128

__device__ __forceinline__ unsigned int orderKey(float f) {
    unsigned int u = __float_as_uint(f);
    return (u & 0x80000000u) ? ~u : (u | 0x80000000u);
}

// ---------------- Kernel Z: zero histograms + scalar slots ----------------
__global__ __launch_bounds__(256) void kZero(unsigned int* __restrict__ hist16,
                                             unsigned int* __restrict__ hist2,
                                             unsigned int* __restrict__ scal) {
    int i = blockIdx.x * 256 + threadIdx.x;
    if (i < 65536) { hist16[i] = 0u; hist2[i] = 0u; }
    if (i < 16) scal[i] = 0u;
}

// ---------------- Kernel A: fused per-row pass (reads the 268MB) ----------
// One wave handles 4 consecutive rows; lane l holds elements 4l..4l+3 (float4).
__global__ __launch_bounds__(256) void kMain(
    const float* __restrict__ y1, const float* __restrict__ y2,
    const int* __restrict__ targets, int N,
    float* __restrict__ totalL, unsigned int* __restrict__ keys,
    float* __restrict__ corrA, unsigned int* __restrict__ maskA,
    unsigned int* __restrict__ hist16,
    float* __restrict__ pKl, float* __restrict__ pTot)
{
    __shared__ float wkl[8], wtot[8];
    const int wave = threadIdx.x >> 5;
    const int lane = threadIdx.x & 31;
    float klAcc = 0.f, totAcc = 0.f;
    const int row0 = (blockIdx.x * 8 + wave) * 4;

    for (int r = 0; r < 4; ++r) {
        const int row = row0 + r;
        if (row >= N) break;
        const float* p1 = y1 + (size_t)row * C_DIM;
        const float* p2 = y2 + (size_t)row * C_DIM;
        if (r < 3) {  // stream next row: emits global_prefetch_b8 on gfx1250
            __builtin_prefetch(p1 + C_DIM + lane * 4, 0, 0);
            __builtin_prefetch(p2 + C_DIM + lane * 4, 0, 0);
        }
        const float4 A4 = ((const float4*)p1)[lane];
        const float4 B4 = ((const float4*)p2)[lane];
        float a[4] = {A4.x, A4.y, A4.z, A4.w};
        float b[4] = {B4.x, B4.y, B4.z, B4.w};

        // argmax/max, matrix 1 (first-index tie-break like jnp.argmax)
        float m1 = a[0]; int i1 = lane * 4;
        #pragma unroll
        for (int j = 1; j < 4; ++j) if (a[j] > m1) { m1 = a[j]; i1 = lane * 4 + j; }
        #pragma unroll
        for (int off = 16; off > 0; off >>= 1) {
            float ov = __shfl_xor(m1, off, 32);
            int   oi = __shfl_xor(i1, off, 32);
            if (ov > m1 || (ov == m1 && oi < i1)) { m1 = ov; i1 = oi; }
        }
        // argmax/max, matrix 2
        float m2 = b[0]; int i2 = lane * 4;
        #pragma unroll
        for (int j = 1; j < 4; ++j) if (b[j] > m2) { m2 = b[j]; i2 = lane * 4 + j; }
        #pragma unroll
        for (int off = 16; off > 0; off >>= 1) {
            float ov = __shfl_xor(m2, off, 32);
            int   oi = __shfl_xor(i2, off, 32);
            if (ov > m2 || (ov == m2 && oi < i2)) { m2 = ov; i2 = oi; }
        }
        // sum-exp
        float e1[4], e2[4], s1 = 0.f, s2 = 0.f;
        #pragma unroll
        for (int j = 0; j < 4; ++j) {
            e1[j] = __expf(a[j] - m1); s1 += e1[j];
            e2[j] = __expf(b[j] - m2); s2 += e2[j];
        }
        #pragma unroll
        for (int off = 16; off > 0; off >>= 1) {
            s1 += __shfl_xor(s1, off, 32);
            s2 += __shfl_xor(s2, off, 32);
        }
        const float lZ1 = __logf(s1), lZ2 = __logf(s2);
        const float inv1 = 1.f / s1, inv2 = 1.f / s2;  // = conf1, conf2

        // CE at target
        const int t = targets[row];
        const int tj = t & 3, tl = t >> 2;
        float av = (tj == 0) ? a[0] : (tj == 1) ? a[1] : (tj == 2) ? a[2] : a[3];
        float bv = (tj == 0) ? b[0] : (tj == 1) ? b[1] : (tj == 2) ? b[2] : b[3];
        const float y1t = __shfl(av, tl, 32);
        const float y2t = __shfl(bv, tl, 32);
        const float total = (m1 + lZ1 - y1t) + (m2 + lZ2 - y2t);

        // symmetric KL row-sum: sum (p1-p2)*(logp1-logp2)
        float kll = 0.f;
        #pragma unroll
        for (int j = 0; j < 4; ++j) {
            const float lp1 = a[j] - m1 - lZ1;
            const float lp2 = b[j] - m2 - lZ2;
            kll += (e1[j] * inv1 - e2[j] * inv2) * (lp1 - lp2);
        }
        #pragma unroll
        for (int off = 16; off > 0; off >>= 1) kll += __shfl_xor(kll, off, 32);

        // pseudo-label correction terms (pl = argmax1)
        const int pj = i1 & 3, plane = i1 >> 2;
        float bvp = (pj == 0) ? b[0] : (pj == 1) ? b[1] : (pj == 2) ? b[2] : b[3];
        const float y2pl = __shfl(bvp, plane, 32);
        const float ce1 = lZ1;                    // -logp1[argmax1]
        const float ce2 = m2 + lZ2 - y2pl;        // -logp2[argmax1]
        const float prod = inv1 * inv2;
        const float corr = sqrtf(prod) * (ce1 + ce2);
        const unsigned int msk = ((i1 == i2) && (prod > 0.5f)) ? 1u : 0u;

        if (lane == 0) {
            const unsigned int key = orderKey(total);
            totalL[row] = total;
            keys[row]   = key;
            corrA[row]  = corr;
            maskA[row]  = msk;
            atomicAdd(&hist16[key >> 16], 1u);
        }
        klAcc += kll; totAcc += total;
    }
    if (lane == 0) { wkl[wave] = klAcc; wtot[wave] = totAcc; }
    __syncthreads();
    if (threadIdx.x == 0) {
        float k = 0.f, tt = 0.f;
        #pragma unroll
        for (int w = 0; w < 8; ++w) { k += wkl[w]; tt += wtot[w]; }
        pKl[blockIdx.x] = k; pTot[blockIdx.x] = tt;
    }
}

// ---------------- Kernel B: find high-16 bin of the k-th smallest ---------
__global__ __launch_bounds__(1024) void kSelect1(const unsigned int* __restrict__ hist,
                                                 unsigned int* __restrict__ scal,
                                                 const int* __restrict__ epochp, int N)
{
    __shared__ unsigned int scan[1024];
    const int t = threadIdx.x;
    if (t == 0) {
        const int epoch = epochp[0];
        const double forget = fmin(0.5, 0.005 * (double)epoch);
        const double rem    = fmax(0.5, 1.0 - forget);
        scal[0] = (unsigned int)((long long)(rem * (double)N));   // num_remember
    }
    unsigned int s = 0;
    for (int j = 0; j < 64; ++j) s += hist[t * 64 + j];
    scan[t] = s;
    __syncthreads();
    for (int off = 1; off < 1024; off <<= 1) {
        const unsigned int add = (t >= off) ? scan[t - off] : 0u;
        __syncthreads();
        scan[t] += add;
        __syncthreads();
    }
    const unsigned int k = scal[0];
    const unsigned int incl = scan[t];
    const unsigned int excl = (t == 0) ? 0u : scan[t - 1];
    if (incl >= k && excl < k) {
        unsigned int cum = excl;
        for (int j = 0; j < 64; ++j) {
            const unsigned int c = hist[t * 64 + j];
            if (cum + c >= k) { scal[1] = (unsigned int)(t * 64 + j); scal[2] = cum; break; }
            cum += c;
        }
    }
}

// ---------------- Kernel C: low-16 histogram inside the threshold bin -----
__global__ __launch_bounds__(256) void kHist2(const unsigned int* __restrict__ keys, int N,
                                              const unsigned int* __restrict__ scal,
                                              unsigned int* __restrict__ hist2)
{
    const int i = blockIdx.x * 256 + threadIdx.x;
    if (i >= N) return;
    const unsigned int key = keys[i];
    if ((key >> 16) == scal[1]) atomicAdd(&hist2[key & 0xFFFFu], 1u);
}

// ---------------- Kernel D: exact 32-bit threshold key + tie count --------
__global__ __launch_bounds__(1024) void kSelect2(const unsigned int* __restrict__ hist,
                                                 unsigned int* __restrict__ scal)
{
    __shared__ unsigned int scan[1024];
    const int t = threadIdx.x;
    unsigned int s = 0;
    for (int j = 0; j < 64; ++j) s += hist[t * 64 + j];
    scan[t] = s;
    __syncthreads();
    for (int off = 1; off < 1024; off <<= 1) {
        const unsigned int add = (t >= off) ? scan[t - off] : 0u;
        __syncthreads();
        scan[t] += add;
        __syncthreads();
    }
    const unsigned int kk = scal[0] - scal[2];  // rank needed inside bin
    const unsigned int incl = scan[t];
    const unsigned int excl = (t == 0) ? 0u : scan[t - 1];
    if (incl >= kk && excl < kk) {
        unsigned int cum = excl;
        for (int j = 0; j < 64; ++j) {
            const unsigned int c = hist[t * 64 + j];
            if (cum + c >= kk) {
                scal[3] = (scal[1] << 16) | (unsigned int)(t * 64 + j);  // threshold key
                scal[4] = kk - cum;                                      // ties to keep clean
                break;
            }
            cum += c;
        }
    }
}

// ---------------- Kernel E: classify clean/noisy, block partials ----------
__global__ __launch_bounds__(256) void kClassify(
    const unsigned int* __restrict__ keys, const float* __restrict__ totalL,
    const float* __restrict__ corrA, const unsigned int* __restrict__ maskA, int N,
    unsigned int* __restrict__ scal,
    float* __restrict__ pClean, float* __restrict__ pCorr, float* __restrict__ pCnt)
{
    __shared__ float rc[256], rs[256], rn[256];
    const int t = threadIdx.x;
    const int i = blockIdx.x * 256 + t;
    float c = 0.f, cs = 0.f, cn = 0.f;
    if (i < N) {
        const unsigned int key = keys[i];
        const unsigned int keyK = scal[3];
        const unsigned int tieTake = scal[4];
        bool clean;
        if (key < keyK) clean = true;
        else if (key == keyK) clean = (atomicAdd(&scal[5], 1u) < tieTake);
        else clean = false;
        if (clean) c = totalL[i];
        else if (maskA[i]) { cs = corrA[i]; cn = 1.f; }
    }
    rc[t] = c; rs[t] = cs; rn[t] = cn;
    __syncthreads();
    for (int off = 128; off > 0; off >>= 1) {
        if (t < off) { rc[t] += rc[t + off]; rs[t] += rs[t + off]; rn[t] += rn[t + off]; }
        __syncthreads();
    }
    if (t == 0) { pClean[blockIdx.x] = rc[0]; pCorr[blockIdx.x] = rs[0]; pCnt[blockIdx.x] = rn[0]; }
}

// ---------------- Kernel F: WMMA-based exact fp32 reductions + finalize ---
// Sum n floats using V_WMMA_F32_16X16X4_F32 with a ones B-matrix:
// D[m,n] += sum_k A[m,k]*1.0  (FMA by 1.0 is exact fp32 addition).
__device__ __forceinline__ float wmmaSum(const float* __restrict__ p, int n, int lane)
{
    float tot;
#if __has_builtin(__builtin_amdgcn_wmma_f32_16x16x4_f32)
    v8f acc = {0.f, 0.f, 0.f, 0.f, 0.f, 0.f, 0.f, 0.f};
    const int chunks = n >> 6;                    // 64 floats per WMMA (16x4 A)
    const int m  = lane & 15;
    const int kb = (lane >> 4) << 1;              // lanes 16-31 hold K=2,3
    v2f ones = {1.f, 1.f};
    for (int c2 = 0; c2 < chunks; ++c2) {
        const float* base = p + c2 * 64 + m * 4 + kb;
        v2f a = {base[0], base[1]};
        acc = __builtin_amdgcn_wmma_f32_16x16x4_f32(
            false, a, false, ones, (short)0, acc, false, false);
    }
    // D: VGPR r = row M=r (lanes 0-15) / M=8+r (lanes 16-31); columns identical.
    float s = acc[0] + acc[1] + acc[2] + acc[3] + acc[4] + acc[5] + acc[6] + acc[7];
    tot = __shfl(s, 0, 32) + __shfl(s, 16, 32);
    for (int i = chunks << 6; i < n; ++i) tot += p[i];
#else
    // Fallback: codegen-confirmed f16 WMMA (16x16x32), 512 floats per step.
    v8f acc = {0.f, 0.f, 0.f, 0.f, 0.f, 0.f, 0.f, 0.f};
    const int chunks = n >> 9;
    const int m = lane & 15;
    const int hi = (lane >> 4) * 8;
    v16h ones;
    #pragma unroll
    for (int j = 0; j < 16; ++j) ones[j] = (_Float16)1.0f;
    for (int c2 = 0; c2 < chunks; ++c2) {
        const float* base = p + c2 * 512 + m * 32;
        v16h a;
        #pragma unroll
        for (int j = 0; j < 16; ++j) {
            const int pr = j >> 1;
            const int k = ((pr < 4) ? 2 * pr : 8 + 2 * pr) + hi + (j & 1);
            a[j] = (_Float16)base[k];
        }
        acc = __builtin_amdgcn_wmma_f32_16x16x32_f16(
            false, a, false, ones, (short)0, acc, false, false);
    }
    float s = acc[0] + acc[1] + acc[2] + acc[3] + acc[4] + acc[5] + acc[6] + acc[7];
    tot = __shfl(s, 0, 32) + __shfl(s, 16, 32);
    for (int i = chunks << 9; i < n; ++i) tot += p[i];
#endif
    return tot;
}

__global__ __launch_bounds__(32) void kFinal(
    const float* __restrict__ pKl, int nA, const float* __restrict__ pTot,
    const float* __restrict__ pClean, int nE, const float* __restrict__ pCorr,
    const float* __restrict__ pCnt, const unsigned int* __restrict__ scal,
    const int* __restrict__ epochp, int N, float* __restrict__ out)
{
    const int lane = threadIdx.x;
    const float klSum    = wmmaSum(pKl,    nA, lane);
    const float totSum   = wmmaSum(pTot,   nA, lane);
    const float cleanSum = wmmaSum(pClean, nE, lane);
    const float corrSum  = wmmaSum(pCorr,  nE, lane);
    const float cntSum   = wmmaSum(pCnt,   nE, lane);
    if (lane == 0) {
        const int epoch = epochp[0];
        float res;
        if (epoch == 0) {
            res = totSum / (float)N;
        } else {
            const float k = (float)scal[0];
            const float clean_loss = cleanSum / k;
            const float corr_mean  = (cntSum > 0.f) ? (corrSum / cntSum) : 0.f;
            const float kl_loss    = klSum / (float)N;
            res = clean_loss + corr_mean + 0.1f * kl_loss;
        }
        out[0] = res;
    }
}

// ---------------- Host launcher -------------------------------------------
extern "C" void kernel_launch(void* const* d_in, const int* in_sizes, int n_in,
                              void* d_out, int out_size, void* d_ws, size_t ws_size,
                              hipStream_t stream)
{
    const float* y1      = (const float*)d_in[0];
    const float* y2      = (const float*)d_in[1];
    const int*   targets = (const int*)d_in[2];
    const int*   epochp  = (const int*)d_in[3];
    const int N = in_sizes[2];

    char* ws = (char*)d_ws;
    const size_t Nsz = (size_t)N;
    float*        totalL = (float*)ws;                           // N f32
    unsigned int* keys   = (unsigned int*)(ws + 4 * Nsz);        // N u32
    float*        corrA  = (float*)(ws + 8 * Nsz);               // N f32
    unsigned int* maskA  = (unsigned int*)(ws + 12 * Nsz);       // N u32
    unsigned int* hist16 = (unsigned int*)(ws + 16 * Nsz);       // 65536 u32
    unsigned int* hist2  = hist16 + 65536;                       // 65536 u32
    unsigned int* scal   = hist2 + 65536;                        // 64 u32
    const int blocksA = (N + 31) / 32;     // 8 waves x 4 rows per block
    const int blocksE = (N + 255) / 256;
    float* pKl   = (float*)(scal + 64);
    float* pTot  = pKl + blocksA;
    float* pClean = pTot + blocksA;
    float* pCorr  = pClean + blocksE;
    float* pCnt   = pCorr + blocksE;

    kZero<<<256, 256, 0, stream>>>(hist16, hist2, scal);
    kMain<<<blocksA, 256, 0, stream>>>(y1, y2, targets, N, totalL, keys, corrA,
                                       maskA, hist16, pKl, pTot);
    kSelect1<<<1, 1024, 0, stream>>>(hist16, scal, epochp, N);
    kHist2<<<(N + 255) / 256, 256, 0, stream>>>(keys, N, scal, hist2);
    kSelect2<<<1, 1024, 0, stream>>>(hist2, scal);
    kClassify<<<blocksE, 256, 0, stream>>>(keys, totalL, corrA, maskA, N, scal,
                                           pClean, pCorr, pCnt);
    kFinal<<<1, 32, 0, stream>>>(pKl, blocksA, pTot, pClean, blocksE, pCorr, pCnt,
                                 scal, epochp, N, (float*)d_out);
}
